// HMultiheadAttention_25434796326977
// MI455X (gfx1250) — matrix operand order repaired and checked by reading, not compiled, not run on hardware
//
#include <hip/hip_runtime.h>
#include <cmath>

typedef __attribute__((ext_vector_type(16))) _Float16 v16h;
typedef __attribute__((ext_vector_type(8)))  float    v8f;
typedef _Float16 half_t;

#define Bc 2
#define Sc 2048
#define Ec 512
#define Hc 8
#define Dc 64
#define MAXN 0.99999f
#define EPSF 1e-6f
#define LOG2E 1.4426950408889634f

__device__ __forceinline__ int lane_id() { return threadIdx.x & 31; }

// ---- WMMA fragment helpers (contiguous k = 16*(lane/16) + s order, wave32) ----

// A fragment: 16x32 f16, row-major source, leading dim ld, tile at (row0, k0)
__device__ __forceinline__ v16h load_a_rm(const half_t* __restrict__ p, int ld, int row0, int k0) {
  int l = lane_id(); int m = l & 15; int g = l >> 4;
  const half_t* r = p + (size_t)(row0 + m) * ld;
  v16h a;
#pragma unroll
  for (int v = 0; v < 8; ++v) {
    int kb = k0 + g * 16 + 2 * v;
    a[2 * v]     = r[kb];
    a[2 * v + 1] = r[kb + 1];
  }
  return a;
}

// B fragment from matrix stored [n][k] row-major (e.g. W[e][c], rows of keys)
__device__ __forceinline__ v16h load_b_nk(const half_t* __restrict__ p, int ld, int n0, int k0) {
  int l = lane_id(); int n = l & 15; int g = l >> 4;
  const half_t* r = p + (size_t)(n0 + n) * ld;
  v16h b;
#pragma unroll
  for (int v = 0; v < 8; ++v) {
    int kb = k0 + g * 16 + 2 * v;
    b[2 * v]     = r[kb];
    b[2 * v + 1] = r[kb + 1];
  }
  return b;
}

// B fragment from matrix stored [k][n] row-major (e.g. augmented V [S][80])
__device__ __forceinline__ v16h load_b_kn(const half_t* __restrict__ p, int ld, int k0, int n0) {
  int l = lane_id(); int n = l & 15; int g = l >> 4;
  v16h b;
#pragma unroll
  for (int v = 0; v < 8; ++v) {
    int kb = k0 + g * 16 + 2 * v;
    b[2 * v]     = p[(size_t)kb * ld + n0 + n];
    b[2 * v + 1] = p[(size_t)(kb + 1) * ld + n0 + n];
  }
  return b;
}

__device__ __forceinline__ v8f wmma16(v16h a, v16h b, v8f c) {
  return __builtin_amdgcn_wmma_f32_16x16x32_f16(false, a, false, b, (short)0, c, false, false);
}

// ---- Kernel 1: f32 -> f16 weight convert ----
__global__ __launch_bounds__(256) void k_cvt16(const float* __restrict__ W, half_t* __restrict__ Wh) {
  int i = blockIdx.x * 256 + threadIdx.x;
  Wh[i] = (half_t)W[i];
}

// ---- Kernel 2: logmap0 per row of 512, emit f16 tangent vectors ----
__global__ __launch_bounds__(256) void k_logmap0(const float* __restrict__ X, half_t* __restrict__ U) {
  __shared__ float red[256];
  int r = blockIdx.x; int tid = threadIdx.x;
  const float* x = X + (size_t)r * Ec;
  float x0 = x[tid], x1 = x[tid + 256];
  red[tid] = x0 * x0 + x1 * x1;
  __syncthreads();
#pragma unroll
  for (int s = 128; s > 0; s >>= 1) {
    if (tid < s) red[tid] += red[tid + s];
    __syncthreads();
  }
  float n = sqrtf(fmaxf(red[0], 1e-12f));
  float f = atanhf(fminf(n, MAXN)) / n;
  U[(size_t)r * Ec + tid]       = (half_t)(x0 * f);
  U[(size_t)r * Ec + tid + 256] = (half_t)(x1 * f);
}

// ---- Kernel 3: T = U @ W^T + bias, f16 WMMA, f32 accumulate ----
__global__ __launch_bounds__(128) void k_gemm(const half_t* __restrict__ A,
                                              const half_t* __restrict__ Wh,
                                              const float* __restrict__ bias,
                                              float* __restrict__ T) {
  int wv = threadIdx.x >> 5;
  int row0 = blockIdx.x * 64 + wv * 16;
  int col0 = blockIdx.y * 64;
  v8f z = {};
  v8f acc[4] = {z, z, z, z};
  for (int k0 = 0; k0 < Ec; k0 += 32) {
    if (k0 + 32 < Ec) {
      __builtin_prefetch(A  + (size_t)row0 * Ec + k0 + 32, 0, 1);
      __builtin_prefetch(Wh + (size_t)col0 * Ec + k0 + 32, 0, 1);
    }
    v16h a = load_a_rm(A, Ec, row0, k0);
#pragma unroll
    for (int nt = 0; nt < 4; ++nt) {
      v16h b = load_b_nk(Wh, Ec, col0 + nt * 16, k0);
      acc[nt] = wmma16(a, b, acc[nt]);
    }
  }
  int l = lane_id(); int n = l & 15; int g = l >> 4;
#pragma unroll
  for (int nt = 0; nt < 4; ++nt)
#pragma unroll
    for (int r = 0; r < 8; ++r) {
      int row = row0 + r + 8 * g;
      int col = col0 + nt * 16 + n;
      T[(size_t)row * Ec + col] = acc[nt][r] + bias[col];
    }
}

// ---- Kernel 4: expmap0 over E, split heads, per-head norms / lambda / V-augment ----
__global__ __launch_bounds__(256) void k_pack(const float* __restrict__ T, int mode,
                                              half_t* __restrict__ QK16, float* __restrict__ N2,
                                              half_t* __restrict__ VAUG) {
  __shared__ float red[8];
  __shared__ float tot;
  int r = blockIdx.x; int b = r / Sc; int s = r % Sc;
  int tid = threadIdx.x; int w = tid >> 5; int lane = tid & 31;
  const float* t = T + (size_t)r * Ec;
  float t0 = t[2 * tid], t1 = t[2 * tid + 1];
  float p = t0 * t0 + t1 * t1;
#pragma unroll
  for (int off = 16; off; off >>= 1) p += __shfl_xor(p, off, 32);
  if (lane == 0) red[w] = p;
  __syncthreads();
  if (tid == 0) { float a = 0.f; for (int i = 0; i < 8; ++i) a += red[i]; tot = a; }
  __syncthreads();
  float n = sqrtf(fmaxf(tot, 1e-12f));
  float f = tanhf(n) / n;                       // expmap0
  float y0 = t0 * f, y1 = t1 * f;
  float hp = y0 * y0 + y1 * y1;                 // per-head norm^2
#pragma unroll
  for (int off = 16; off; off >>= 1) hp += __shfl_xor(hp, off, 32);
  int bh = b * Hc + w;
  int d0 = 2 * lane;
  if (mode < 2) {                               // q or k
    if (lane == 0) N2[(size_t)bh * Sc + s] = hp;
    half_t* q = QK16 + ((size_t)bh * Sc + s) * Dc;
    q[d0] = (half_t)y0; q[d0 + 1] = (half_t)y1;
  } else {                                      // v: augmented row of 80
    float lam = 2.0f / fmaxf(1.0f - hp, EPSF);
    half_t* va = VAUG + ((size_t)bh * Sc + s) * 80;
    va[d0] = (half_t)(y0 * lam); va[d0 + 1] = (half_t)(y1 * lam);
    if (lane == 0) va[64] = (half_t)(lam - 1.0f);
    if (lane >= 1 && lane < 16) va[64 + lane] = (half_t)0.0f;
  }
}

// ---- Kernel 5: streaming causal hyperbolic attention (flash-style) ----
__global__ __launch_bounds__(128) void k_attn(const half_t* __restrict__ QH,
                                              const half_t* __restrict__ KH,
                                              const half_t* __restrict__ VA,
                                              const float* __restrict__ QN2,
                                              const float* __restrict__ KN2,
                                              float* __restrict__ NUM,
                                              float* __restrict__ DEN,
                                              const float* __restrict__ tau,
                                              const float* __restrict__ gam) {
  __shared__ half_t wtile[4][16][32];
  int wv = threadIdx.x >> 5;
  int bh = blockIdx.y;
  int i0 = blockIdx.x * 64 + wv * 16;
  int lane = lane_id(); int n = lane & 15; int g = lane >> 4;

  const half_t* Q = QH + (size_t)bh * Sc * Dc;
  const half_t* K = KH + (size_t)bh * Sc * Dc;
  const half_t* V = VA + (size_t)bh * Sc * 80;
  const float* qn2 = QN2 + (size_t)bh * Sc;
  const float* kn2 = KN2 + (size_t)bh * Sc;

  // w = exp(gamma - e^tau * acosh(z)) = exp2(gamma*log2e - e^tau * log2(z + sqrt(z^2-1)))
  float etau = __builtin_amdgcn_exp2f(tau[0] * LOG2E);
  float g2   = gam[0] * LOG2E;

  v16h qa0 = load_a_rm(Q, Dc, i0, 0);
  v16h qa1 = load_a_rm(Q, Dc, i0, 32);

  float q2[8], om_q2[8];
#pragma unroll
  for (int r = 0; r < 8; ++r) { q2[r] = qn2[i0 + r + 8 * g]; om_q2[r] = 1.0f - q2[r]; }

  v8f z = {};
  v8f acc[5] = {z, z, z, z, z};

  int jmax = i0 + 15;
  for (int j0 = 0; j0 <= jmax; j0 += 32) {
#pragma unroll
    for (int tt = 0; tt < 2; ++tt) {
      int jt = j0 + tt * 16;
      if (jt <= jmax) {                          // wave-uniform
        v8f c = {};
        v16h b0 = load_b_nk(K, Dc, jt, 0);
        v16h b1 = load_b_nk(K, Dc, jt, 32);
        c = wmma16(qa0, b0, c);
        c = wmma16(qa1, b1, c);
        int j = jt + n;
        float k2 = kn2[j];
        float om_k2 = 1.0f - k2;
#pragma unroll
        for (int r = 0; r < 8; ++r) {
          int i = i0 + r + 8 * g;
          float w = 0.0f;
          if (j <= i) {
            float sq  = fmaxf(q2[r] + k2 - 2.0f * c[r], 0.0f);
            float dn  = fmaxf(om_q2[r] * om_k2, EPSF);
            float arg = fmaxf(fmaf(2.0f * sq, __builtin_amdgcn_rcpf(dn), 1.0f),
                              1.0f + 1e-7f);
            float zz  = arg + __builtin_amdgcn_sqrtf(fmaf(arg, arg, -1.0f));
            w = __builtin_amdgcn_exp2f(g2 - etau * __builtin_amdgcn_logf(zz));
          }
          wtile[wv][r + 8 * g][tt * 16 + n] = (half_t)w;
        }
      } else {
#pragma unroll
        for (int r = 0; r < 8; ++r) wtile[wv][r + 8 * g][tt * 16 + n] = (half_t)0.0f;
      }
    }
    // re-layout 16x32 score tile (D-frag -> A-frag) through wave-private LDS
    v16h wf = load_a_rm(&wtile[wv][0][0], 32, 0, 0);
    const half_t* vrow = V + (size_t)j0 * 80;
#pragma unroll
    for (int nt = 0; nt < 5; ++nt) {
      v16h bv = load_b_kn(vrow, 80, 0, nt * 16);
      acc[nt] = wmma16(wf, bv, acc[nt]);
    }
  }

#pragma unroll
  for (int nt = 0; nt < 5; ++nt)
#pragma unroll
    for (int r = 0; r < 8; ++r) {
      int i = i0 + r + 8 * g;
      int col = nt * 16 + n;
      if (col < Dc)       NUM[((size_t)bh * Sc + i) * Dc + col] = acc[nt][r];
      else if (col == Dc) DEN[(size_t)bh * Sc + i] = acc[nt][r];
    }
}

// ---- Kernel 6: gyromidpoint + beta-concat + final expmap0 ----
__global__ __launch_bounds__(256) void k_final(const float* __restrict__ NUM,
                                               const float* __restrict__ DEN,
                                               float* __restrict__ OUT, float beta) {
  __shared__ float red[8];
  __shared__ float tot;
  int r = blockIdx.x; int b = r / Sc; int s = r % Sc;
  int tid = threadIdx.x; int w = tid >> 5; int lane = tid & 31;
  int bh = b * Hc + w;
  const float* num = NUM + ((size_t)bh * Sc + s) * Dc;
  float den = fmaxf(DEN[(size_t)bh * Sc + s], EPSF);
  int d0 = 2 * lane;
  float m0 = num[d0] / den, m1 = num[d0 + 1] / den;
  float p = m0 * m0 + m1 * m1;
#pragma unroll
  for (int off = 16; off; off >>= 1) p += __shfl_xor(p, off, 32);
  float mn = sqrtf(fmaxf(p, 1e-12f));
  float midf = tanhf(0.5f * atanhf(fminf(mn, MAXN))) / mn;       // gyromidpoint scale
  float midn = sqrtf(fmaxf(midf * mn * midf * mn, 1e-12f));
  float tf = atanhf(fminf(midn, MAXN)) / midn * midf * beta;     // logmap0 * beta
  float t0 = m0 * tf, t1 = m1 * tf;
  float pp = t0 * t0 + t1 * t1;
#pragma unroll
  for (int off = 16; off; off >>= 1) pp += __shfl_xor(pp, off, 32);
  if (lane == 0) red[w] = pp;
  __syncthreads();
  if (tid == 0) { float a = 0.f; for (int i = 0; i < 8; ++i) a += red[i]; tot = a; }
  __syncthreads();
  float nn = sqrtf(fmaxf(tot, 1e-12f));
  float f = tanhf(nn) / nn;                                      // final expmap0
  float* o = OUT + (size_t)r * Ec;
  o[w * Dc + d0] = t0 * f;
  o[w * Dc + d0 + 1] = t1 * f;
}

extern "C" void kernel_launch(void* const* d_in, const int* in_sizes, int n_in,
                              void* d_out, int out_size, void* d_ws, size_t ws_size,
                              hipStream_t stream) {
  const float* X[3]  = {(const float*)d_in[0], (const float*)d_in[1], (const float*)d_in[2]};
  const float* W[3]  = {(const float*)d_in[3], (const float*)d_in[4], (const float*)d_in[5]};
  const float* BI[3] = {(const float*)d_in[6], (const float*)d_in[7], (const float*)d_in[8]};
  const float* tau = (const float*)d_in[9];
  const float* gam = (const float*)d_in[10];

  char* ws = (char*)d_ws;
  size_t off = 0;
  auto carve = [&](size_t bytes) { void* p = ws + off; off += (bytes + 255) & ~(size_t)255; return p; };

  half_t* uh   = (half_t*)carve((size_t)Bc * Sc * Ec * 2);        // tangent activations f16
  half_t* Wh   = (half_t*)carve((size_t)Ec * Ec * 2);             // f16 weights
  float*  tbuf = (float*) carve((size_t)Bc * Sc * Ec * 4);        // GEMM output f32
  half_t* qh   = (half_t*)carve((size_t)Bc * Hc * Sc * Dc * 2);
  half_t* kh   = (half_t*)carve((size_t)Bc * Hc * Sc * Dc * 2);
  half_t* vaug = (half_t*)carve((size_t)Bc * Hc * Sc * 80 * 2);
  float*  qn2  = (float*) carve((size_t)Bc * Hc * Sc * 4);
  float*  kn2  = (float*) carve((size_t)Bc * Hc * Sc * 4);
  float*  numb = (float*) carve((size_t)Bc * Hc * Sc * Dc * 4);
  float*  den  = (float*) carve((size_t)Bc * Hc * Sc * 4);

  // beta-concatenation ratio: exp(lbeta(E/2,1/2) - lbeta(D/2,1/2))
  double lb1 = lgamma(256.0) + lgamma(0.5) - lgamma(256.5);
  double lb2 = lgamma(32.0)  + lgamma(0.5) - lgamma(32.5);
  float beta = (float)exp(lb1 - lb2);

  for (int m = 0; m < 3; ++m) {
    k_cvt16<<<(Ec * Ec) / 256, 256, 0, stream>>>(W[m], Wh);
    k_logmap0<<<Bc * Sc, 256, 0, stream>>>(X[m], uh);
    k_gemm<<<dim3(Bc * Sc / 64, Ec / 64), 128, 0, stream>>>(uh, Wh, BI[m], tbuf);
    half_t* qk = (m == 0) ? qh : kh;
    float*  n2 = (m == 0) ? qn2 : kn2;
    k_pack<<<Bc * Sc, 256, 0, stream>>>(tbuf, m, qk, n2, vaug);
  }

  k_attn<<<dim3(Sc / 64, Bc * Hc), 128, 0, stream>>>(qh, kh, vaug, qn2, kn2, numb, den, tau, gam);
  k_final<<<Bc * Sc, 256, 0, stream>>>(numb, den, (float*)d_out, beta);
}
